// EncoderBlock_88484916232285
// MI455X (gfx1250) — matrix operand order repaired and checked by reading, not compiled, not run on hardware
//
#include <hip/hip_runtime.h>
#include <hip/hip_bf16.h>
#include <math.h>

#define BDIM 256

typedef __attribute__((ext_vector_type(16))) __bf16 v16bf;
typedef __attribute__((ext_vector_type(8)))  float  v8f;
typedef __attribute__((ext_vector_type(4)))  int    v4i;

typedef __attribute__((address_space(1))) v4i g_v4i;   // global v4i32
typedef __attribute__((address_space(3))) v4i l_v4i;   // LDS v4i32

// ---- CDNA5 async global->LDS staging (guarded; falls back to uint4 copy) ----
#if defined(__has_builtin)
#  if __has_builtin(__builtin_amdgcn_global_load_async_to_lds_b128)
#    define HAVE_ASYNC_LDS 1
#    define ASYNC_B128(g, l) \
       __builtin_amdgcn_global_load_async_to_lds_b128((g_v4i*)(g), (l_v4i*)(l), 0, 0)
#  endif
#endif
#ifndef HAVE_ASYNC_LDS
#  define HAVE_ASYNC_LDS 0
#endif

__device__ __forceinline__ void copy16(const __bf16* g, __bf16* l) {
#if HAVE_ASYNC_LDS
  ASYNC_B128(g, l);
#else
  *(uint4*)l = *(const uint4*)g;
#endif
}

__device__ __forceinline__ void wait_async0() {
#if HAVE_ASYNC_LDS
#  if __has_builtin(__builtin_amdgcn_s_wait_asynccnt)
  __builtin_amdgcn_s_wait_asynccnt(0);
#  else
  asm volatile("s_wait_asynccnt 0x0" ::: "memory");
#  endif
#endif
}

union FragU { v16bf v; unsigned u[8]; };

__device__ __forceinline__ v8f zero8() {
  v8f z;
#pragma unroll
  for (int i = 0; i < 8; ++i) z[i] = 0.f;
  return z;
}

// A fragment (16x32 bf16): lane r=lane&15 -> row; half hh=lane>>4;
// vgpr v in 0..3: k = hh*8 + 2v{+1}; v in 4..7: k = 16 + hh*8 + 2(v-4){+1}.
__device__ __forceinline__ v16bf lds_frag_a(const __bf16* base, int stride, int lane) {
  const int hh = lane >> 4, r = lane & 15;
  const unsigned* p = (const unsigned*)(base + r * stride);
  FragU f;
#pragma unroll
  for (int v = 0; v < 4; ++v) {
    f.u[v]     = p[hh * 4 + v];
    f.u[4 + v] = p[8 + hh * 4 + v];
  }
  return f.v;
}

// B fragment (32x16 bf16) from TRANSPOSED tile Bt[n][k]:
// lane n=lane&15 -> col; vgpr v in 0..3: k = hh*8 + 2v{+1}; v in 4..7: k = 16 + hh*8 + 2(v-4){+1}.
__device__ __forceinline__ v16bf lds_frag_b(const __bf16* base, int stride, int lane) {
  const int hh = lane >> 4, n = lane & 15;
  const unsigned* p = (const unsigned*)(base + n * stride);
  FragU f;
#pragma unroll
  for (int v = 0; v < 4; ++v) {
    f.u[v]     = p[hh * 4 + v];
    f.u[4 + v] = p[8 + hh * 4 + v];
  }
  return f.v;
}

__device__ __forceinline__ v8f wmma_bf16(v16bf a, v16bf b, v8f c) {
  return __builtin_amdgcn_wmma_f32_16x16x32_bf16(false, a, false, b, (short)0, c,
                                                 false, false);
}

// ---------------------------------------------------------------------------
// 128x128-tile bf16 GEMM, double-buffered LDS, async staging.
// A[M,K] row-major; Bt[N,K] = B transposed (weights pre-transposed).
// EPI 0: q/k -> bf16 qkbuf[m][1536] (cols<768 scaled 1/8); v -> vT[b,h,dh,s]
// EPI 1: f32 out = acc + resid
// EPI 2: bf16 out = gelu(acc + bias)
// EPI 3: f32 out = acc + bias + resid
// ---------------------------------------------------------------------------
template <int EPI>
__global__ __launch_bounds__(BDIM)
void gemm_kernel(const __bf16* __restrict__ A, const __bf16* __restrict__ Bt,
                 float* __restrict__ Cf, __bf16* __restrict__ Cb,
                 __bf16* __restrict__ Cv,
                 const float* __restrict__ bias, const float* __restrict__ resid,
                 int N, int K)
{
  constexpr int AST = 40;
  __shared__ __bf16 As[2][128 * AST];   // [m][k]
  __shared__ __bf16 Bs[2][128 * AST];   // [n][k]

  const int tid  = threadIdx.x;
  const int lane = tid & 31;
  const int w    = tid >> 5;
  const int wm   = w >> 2;              // 0..1
  const int wn   = w & 3;               // 0..3
  const int m0   = blockIdx.y * 128;
  const int n0   = blockIdx.x * 128;

  const int lr = tid >> 1;              // tile row 0..127
  const int lc = (tid & 1) * 16;        // tile col seg (16 bf16 = 2 x b128)

  v8f acc[4][2];
#pragma unroll
  for (int i = 0; i < 4; ++i)
#pragma unroll
    for (int j = 0; j < 2; ++j) acc[i][j] = zero8();

  auto stage = [&](int buf, int k0) {
    const __bf16* ga = A  + (size_t)(m0 + lr) * K + k0 + lc;
    copy16(ga,     &As[buf][lr * AST + lc]);
    copy16(ga + 8, &As[buf][lr * AST + lc + 8]);
    const __bf16* gb = Bt + (size_t)(n0 + lr) * K + k0 + lc;
    copy16(gb,     &Bs[buf][lr * AST + lc]);
    copy16(gb + 8, &Bs[buf][lr * AST + lc + 8]);
  };

  const int nk = K / 32;
  stage(0, 0);
  wait_async0();
  __syncthreads();

  for (int t = 0; t < nk; ++t) {
    if (t + 1 < nk) stage((t + 1) & 1, (t + 1) * 32);

    const __bf16* Ab = &As[t & 1][0];
    const __bf16* Bb = &Bs[t & 1][0];
    v16bf af[4], bfr[2];
#pragma unroll
    for (int mt = 0; mt < 4; ++mt)
      af[mt] = lds_frag_a(Ab + (wm * 64 + mt * 16) * AST, AST, lane);
#pragma unroll
    for (int nt = 0; nt < 2; ++nt)
      bfr[nt] = lds_frag_b(Bb + (wn * 32 + nt * 16) * AST, AST, lane);
#pragma unroll
    for (int mt = 0; mt < 4; ++mt)
#pragma unroll
      for (int nt = 0; nt < 2; ++nt)
        acc[mt][nt] = wmma_bf16(af[mt], bfr[nt], acc[mt][nt]);

    wait_async0();
    __syncthreads();
  }

  // C layout: m = vgpr + 8*(lane>=16), n = lane&15
  const int cn    = n0 + wn * 32 + (lane & 15);
  const int mbase = m0 + wm * 64 + ((lane >> 4) << 3);
#pragma unroll
  for (int mt = 0; mt < 4; ++mt) {
#pragma unroll
    for (int nt = 0; nt < 2; ++nt) {
      const int n = cn + nt * 16;
#pragma unroll
      for (int v = 0; v < 8; ++v) {
        const int m = mbase + mt * 16 + v;
        float val = acc[mt][nt][v];
        if (EPI == 0) {
          if (n < 1536) {
            if (n < 768) val *= 0.125f;          // 1/sqrt(64) folded into q
            Cb[(size_t)m * 1536 + n] = (__bf16)val;
          } else {
            const int hh = (n - 1536) >> 6, e = (n - 1536) & 63;
            const int bb = m >> 10, ss = m & 1023;
            Cv[(((size_t)bb * 12 + hh) * 64 + e) * 1024 + ss] = (__bf16)val;
          }
        } else if (EPI == 1) {
          const size_t idx = (size_t)m * N + n;
          Cf[idx] = val + resid[idx];
        } else if (EPI == 2) {
          const float xv = val + bias[n];
          Cb[(size_t)m * N + n] =
              (__bf16)(0.5f * xv * (1.f + erff(xv * 0.70710678118f)));
        } else {
          const size_t idx = (size_t)m * N + n;
          Cf[idx] = val + bias[n] + resid[idx];
        }
      }
    }
  }
}

// ---------------------------------------------------------------------------
// Fused flash attention. qk: [B*S, 1536] bf16 (q | k, per-head 64 cols,
// q pre-scaled). vT: [B*H, 64, 1024] bf16 (dh-major). Grid (S/128, B*H).
// Wave w owns q rows [w*16, w*16+16). Double-buffered K/V chunks of 64.
// o_out: [B*S, 768] bf16 head-concat.
// ---------------------------------------------------------------------------
__global__ __launch_bounds__(BDIM)
void attn_kernel(const __bf16* __restrict__ qk, const __bf16* __restrict__ vT,
                 __bf16* __restrict__ o_out)
{
  constexpr int QST = 72;
  __shared__ __bf16 Qs[128 * QST];        // [qrow][dh]
  __shared__ __bf16 Ks[2][64 * QST];      // [krow][dh]  (transposed-B for scores)
  __shared__ __bf16 Vt[2][64 * QST];      // [dh][krow]  (transposed-B for P x V)
  __shared__ __bf16 Ps[8][16 * QST];      // per-wave P scratch [m][krow]

  const int tid  = threadIdx.x;
  const int lane = tid & 31;
  const int w    = tid >> 5;
  const int b    = blockIdx.y / 12;
  const int h    = blockIdx.y % 12;
  const int q0   = blockIdx.x * 128;
  const size_t rowbase = (size_t)b * 1024;
  const __bf16* vhead  = vT + (size_t)blockIdx.y * 64 * 1024;

  {   // Q block 128x64, async
    const int r = tid >> 1;
    const int c = (tid & 1) * 32;
    const __bf16* src = qk + (rowbase + q0 + r) * 1536 + h * 64 + c;
    __bf16* dst = Qs + r * QST + c;
    copy16(src,      dst);
    copy16(src + 8,  dst + 8);
    copy16(src + 16, dst + 16);
    copy16(src + 24, dst + 24);
  }

  const int kr = tid >> 2;               // 0..63
  const int kcseg = (tid & 3) * 16;      // 0..48
  auto stageKV = [&](int buf, int kc) {
    const __bf16* ksrc = qk + (rowbase + kc + kr) * 1536 + 768 + h * 64 + kcseg;
    copy16(ksrc,     &Ks[buf][kr * QST + kcseg]);
    copy16(ksrc + 8, &Ks[buf][kr * QST + kcseg + 8]);
    const __bf16* vsrc = vhead + (size_t)kr * 1024 + kc + kcseg;   // row = dh
    copy16(vsrc,     &Vt[buf][kr * QST + kcseg]);
    copy16(vsrc + 8, &Vt[buf][kr * QST + kcseg + 8]);
  };

  stageKV(0, 0);
  wait_async0();
  __syncthreads();

  v16bf qa[2];
  qa[0] = lds_frag_a(Qs + (w * 16) * QST,      QST, lane);
  qa[1] = lds_frag_a(Qs + (w * 16) * QST + 32, QST, lane);

  float rowmax[8], rowsum[8];
  v8f oacc[4];
#pragma unroll
  for (int v = 0; v < 8; ++v) { rowmax[v] = -3.0e38f; rowsum[v] = 0.f; }
#pragma unroll
  for (int nt = 0; nt < 4; ++nt) oacc[nt] = zero8();

  __bf16* Pw = &Ps[w][0];

  for (int t = 0; t < 16; ++t) {
    if (t + 1 < 16) stageKV((t + 1) & 1, (t + 1) * 64);
    const __bf16* Kb = &Ks[t & 1][0];
    const __bf16* Vb = &Vt[t & 1][0];

    // scores: 16 q-rows x 64 k-rows, K = DH = 64
    v8f sc[4];
#pragma unroll
    for (int nt = 0; nt < 4; ++nt) {
      sc[nt] = zero8();
#pragma unroll
      for (int kk = 0; kk < 2; ++kk) {
        v16bf kb = lds_frag_b(Kb + (nt * 16) * QST + kk * 32, QST, lane);
        sc[nt] = wmma_bf16(qa[kk], kb, sc[nt]);
      }
    }

    // online softmax over the 16-lane C-layout row groups
#pragma unroll
    for (int v = 0; v < 8; ++v) {
      float mx = fmaxf(fmaxf(sc[0][v], sc[1][v]), fmaxf(sc[2][v], sc[3][v]));
#pragma unroll
      for (int off = 8; off >= 1; off >>= 1)
        mx = fmaxf(mx, __shfl_xor(mx, off, 32));
      const float nm   = fmaxf(rowmax[v], mx);
      const float corr = __expf(rowmax[v] - nm);
      rowmax[v] = nm;
      float s = 0.f;
      const int m = v + ((lane >> 4) << 3);
#pragma unroll
      for (int nt = 0; nt < 4; ++nt) {
        const float e = __expf(sc[nt][v] - nm);
        s += e;
        Pw[m * QST + nt * 16 + (lane & 15)] = (__bf16)e;
        oacc[nt][v] *= corr;
      }
#pragma unroll
      for (int off = 8; off >= 1; off >>= 1)
        s += __shfl_xor(s, off, 32);
      rowsum[v] = rowsum[v] * corr + s;
    }

    // O += P[16,64] x V[64,64]
#pragma unroll
    for (int kk = 0; kk < 2; ++kk) {
      v16bf pa = lds_frag_a(Pw + kk * 32, QST, lane);
#pragma unroll
      for (int nt = 0; nt < 4; ++nt) {
        v16bf vb = lds_frag_b(Vb + (nt * 16) * QST + kk * 32, QST, lane);
        oacc[nt] = wmma_bf16(pa, vb, oacc[nt]);
      }
    }

    wait_async0();
    __syncthreads();
  }

  // normalize + store head-concat bf16
#pragma unroll
  for (int v = 0; v < 8; ++v) {
    const int m = v + ((lane >> 4) << 3);
    const size_t row = rowbase + q0 + w * 16 + m;
    const float inv = 1.f / rowsum[v];
#pragma unroll
    for (int nt = 0; nt < 4; ++nt)
      o_out[row * 768 + h * 64 + nt * 16 + (lane & 15)] = (__bf16)(oacc[nt][v] * inv);
  }
}

// ---------------------------------------------------------------------------
// Row LayerNorm: (x-mean)/(std+eps)*scale+shift, one block per 768-elem row.
// ---------------------------------------------------------------------------
__global__ __launch_bounds__(BDIM)
void norm_kernel(const float* __restrict__ X, const float* __restrict__ scale,
                 const float* __restrict__ shift, __bf16* __restrict__ out)
{
  __shared__ float red[BDIM];
  const int row = blockIdx.x;
  const int tid = threadIdx.x;
  const float* xr = X + (size_t)row * 768;
  const float x0 = xr[tid], x1 = xr[tid + 256], x2 = xr[tid + 512];
  red[tid] = x0 + x1 + x2;
  __syncthreads();
  for (int off = 128; off >= 1; off >>= 1) {
    if (tid < off) red[tid] += red[tid + off];
    __syncthreads();
  }
  const float mean = red[0] * (1.f / 768.f);
  __syncthreads();
  const float d0 = x0 - mean, d1 = x1 - mean, d2 = x2 - mean;
  red[tid] = d0 * d0 + d1 * d1 + d2 * d2;
  __syncthreads();
  for (int off = 128; off >= 1; off >>= 1) {
    if (tid < off) red[tid] += red[tid + off];
    __syncthreads();
  }
  const float inv = 1.f / (sqrtf(red[0] * (1.f / 768.f)) + 1e-8f);
  __bf16* orow = out + (size_t)row * 768;
  orow[tid]       = (__bf16)(d0 * inv * scale[tid]       + shift[tid]);
  orow[tid + 256] = (__bf16)(d1 * inv * scale[tid + 256] + shift[tid + 256]);
  orow[tid + 512] = (__bf16)(d2 * inv * scale[tid + 512] + shift[tid + 512]);
}

// Pack per-head Wq/Wk/Wv [H,D,DH] fp32 -> one TRANSPOSED bf16 [2304][768].
__global__ void pack_qkv_t_kernel(const float* __restrict__ Wq,
                                  const float* __restrict__ Wk,
                                  const float* __restrict__ Wv,
                                  __bf16* __restrict__ out)
{
  const int idx = blockIdx.x * blockDim.x + threadIdx.x;
  if (idx >= 2304 * 768) return;
  const int n = idx / 768;             // output col (q|k|v, head-major)
  const int d = idx % 768;             // input dim
  const int sel = n / 768;
  const int he  = n % 768;
  const int h = he >> 6, e = he & 63;
  const float* W = (sel == 0) ? Wq : (sel == 1) ? Wk : Wv;
  out[idx] = (__bf16)W[((size_t)h * 768 + d) * 64 + e];
}

// fp32 [K,N] -> bf16 transposed [N,K]
__global__ void f2bf_t_kernel(const float* __restrict__ in, __bf16* __restrict__ out,
                              int K, int N)
{
  const int idx = blockIdx.x * blockDim.x + threadIdx.x;
  if (idx >= K * N) return;
  const int n = idx / K;
  const int k = idx % K;
  out[idx] = (__bf16)in[(size_t)k * N + n];
}

extern "C" void kernel_launch(void* const* d_in, const int* in_sizes, int n_in,
                              void* d_out, int out_size, void* d_ws, size_t ws_size,
                              hipStream_t stream)
{
  (void)in_sizes; (void)n_in; (void)out_size; (void)ws_size;
  const float* x      = (const float*)d_in[0];
  const float* scale1 = (const float*)d_in[1];
  const float* shift1 = (const float*)d_in[2];
  const float* scale2 = (const float*)d_in[3];
  const float* shift2 = (const float*)d_in[4];
  const float* Wq     = (const float*)d_in[5];
  const float* Wk     = (const float*)d_in[6];
  const float* Wv     = (const float*)d_in[7];
  const float* Wo     = (const float*)d_in[8];
  const float* W1     = (const float*)d_in[9];
  const float* b1     = (const float*)d_in[10];
  const float* W2     = (const float*)d_in[11];
  const float* b2     = (const float*)d_in[12];

  const size_t M = 8192;                 // B*S
  char* ws = (char*)d_ws;
  size_t off = 0;
  auto alloc = [&](size_t bytes) -> char* {
    char* p = ws + off;
    off = (off + bytes + 255) & ~(size_t)255;
    return p;
  };
  __bf16* hbf     = (__bf16*)alloc(M * 768 * 2);           // norm1 out
  __bf16* wqkv_t  = (__bf16*)alloc((size_t)2304 * 768 * 2);
  __bf16* wo_t    = (__bf16*)alloc((size_t)768 * 768 * 2);
  __bf16* w1_t    = (__bf16*)alloc((size_t)3072 * 768 * 2);
  __bf16* w2_t    = (__bf16*)alloc((size_t)768 * 3072 * 2);
  __bf16* qkbuf   = (__bf16*)alloc(M * 1536 * 2);          // q|k (q pre-scaled)
  __bf16* vT      = (__bf16*)alloc((size_t)96 * 64 * 1024 * 2);  // [b,h,dh,s]
  __bf16* ocat    = (__bf16*)alloc(M * 768 * 2);           // attn out, head concat
  float*  attn_o  = (float*)alloc(M * 768 * 4);            // + residual1 (fp32)
  __bf16* h2      = (__bf16*)alloc(M * 768 * 2);           // norm2 out
  __bf16* m1      = (__bf16*)alloc(M * 3072 * 2);          // gelu(fc1)

  pack_qkv_t_kernel<<<(2304 * 768 + 255) / 256, 256, 0, stream>>>(Wq, Wk, Wv, wqkv_t);
  f2bf_t_kernel<<<(768 * 768 + 255) / 256, 256, 0, stream>>>(Wo, wo_t, 768, 768);
  f2bf_t_kernel<<<(3072 * 768 + 255) / 256, 256, 0, stream>>>(W1, w1_t, 768, 3072);
  f2bf_t_kernel<<<(768 * 3072 + 255) / 256, 256, 0, stream>>>(W2, w2_t, 3072, 768);

  norm_kernel<<<M, 256, 0, stream>>>(x, scale1, shift1, hbf);
  gemm_kernel<0><<<dim3(2304 / 128, M / 128), 256, 0, stream>>>(
      hbf, wqkv_t, nullptr, qkbuf, vT, nullptr, nullptr, 2304, 768);
  attn_kernel<<<dim3(1024 / 128, 8 * 12), 256, 0, stream>>>(qkbuf, vT, ocat);
  gemm_kernel<1><<<dim3(768 / 128, M / 128), 256, 0, stream>>>(
      ocat, wo_t, attn_o, nullptr, nullptr, nullptr, x, 768, 768);
  norm_kernel<<<M, 256, 0, stream>>>(attn_o, scale2, shift2, h2);
  gemm_kernel<2><<<dim3(3072 / 128, M / 128), 256, 0, stream>>>(
      h2, w1_t, nullptr, m1, nullptr, b1, nullptr, 3072, 768);
  gemm_kernel<3><<<dim3(768 / 128, M / 128), 256, 0, stream>>>(
      m1, w2_t, (float*)d_out, nullptr, nullptr, b2, attn_o, 768, 3072);
}
